// NodeDeepOnet_82265803588247
// MI455X (gfx1250) — matrix operand order repaired and checked by reading, not compiled, not run on hardware
//
#include <hip/hip_runtime.h>

typedef __attribute__((ext_vector_type(16))) _Float16 v16h;
typedef __attribute__((ext_vector_type(8)))  _Float16 v8h;
typedef __attribute__((ext_vector_type(8)))  float    v8f;

#define NK_  262144
#define NC_  262144
#define BSEG 64
#define HID  256
#define MT   64      // rows per block
#define LDA  272     // padded LDS row stride (halves); 544B = 16B-aligned rows

// One 256->256 layer for a wave: A-panel from LDS (CDNA5 16-bit A layout),
// B from global f16ᵀ weights (per-lane contiguous column), zero-initialized
// fp32 accumulators (inline-0 C on first WMMA), results left in accs[8].
// All addresses are a loop-invariant lane base + compile-time constants so the
// compiler folds them into ds_load/global_load immediate offsets.
__device__ __forceinline__ void layer_accs(const _Float16* __restrict__ buf,
                                           const _Float16* __restrict__ wt,
                                           int mtile, int nbase, int lane,
                                           v8f* accs) {
  const int m    = lane & 15;
  const int aoff = (lane & 16) ? 8 : 0;   // A: lanes16-31 hold K+8.. / K+24..
  const int boff = (lane & 16) ? 8 : 0;   // B: lanes16-31 hold K+8.. / K+24..
  const _Float16* abase = buf + (mtile * 16 + m) * LDA + aoff;
  const _Float16* wbase = wt + (nbase + m) * HID + boff;
  v16h afr[8];
#pragma unroll
  for (int kc = 0; kc < 8; ++kc) {
    union { v16h v; v8h h[2]; } u;
    u.h[0] = *(const v8h*)(abase + kc * 32);
    u.h[1] = *(const v8h*)(abase + kc * 32 + 16);
    afr[kc] = u.v;
  }
#pragma unroll
  for (int nt = 0; nt < 8; ++nt) {
    const _Float16* wp = wbase + nt * 16 * HID;
    v8f acc = {};  // first WMMA consumes inline constant 0 as C
#pragma unroll
    for (int kc = 0; kc < 8; ++kc) {
      union { v16h v; v8h h[2]; } u;
      u.h[0] = *(const v8h*)(wp + kc * 32);
      u.h[1] = *(const v8h*)(wp + kc * 32 + 16);
      acc = __builtin_amdgcn_wmma_f32_16x16x32_f16(false, afr[kc], false, u.v,
                                                   (short)0, acc, false, false);
    }
    accs[nt] = acc;
  }
}

// Epilogue bias add (dual-issueable v_add chain instead of pre-WMMA mov chain).
__device__ __forceinline__ void add_bias(v8f* accs, const float* __restrict__ bias,
                                         int nbase, int lane) {
  const int n0 = nbase + (lane & 15);
#pragma unroll
  for (int nt = 0; nt < 8; ++nt) {
    const float bv = bias[n0 + nt * 16];
#pragma unroll
    for (int r = 0; r < 8; ++r) accs[nt][r] += bv;
  }
}

// C/D layout: VGPR r, lanes0-15 -> (M=r, N=lane); lanes16-31 -> (M=r+8, N=lane-16)
__device__ __forceinline__ void store_accs(_Float16* __restrict__ buf,
                                           const v8f* accs, int mtile, int nbase,
                                           int lane, bool relu) {
  const int n0 = nbase + (lane & 15);
  const int rb = mtile * 16 + ((lane & 16) ? 8 : 0);
#pragma unroll
  for (int nt = 0; nt < 8; ++nt) {
    const int n = n0 + nt * 16;
#pragma unroll
    for (int r = 0; r < 8; ++r) {
      float v = accs[nt][r];
      if (relu) v = fmaxf(v, 0.0f);
      buf[(rb + r) * LDA + n] = (_Float16)v;
    }
  }
}

// ---------- small prep kernels ----------
__global__ void k_zero(float* segsum, int* cnt) {
  int tid = blockIdx.x * 256 + threadIdx.x;
  if (tid < BSEG * HID) segsum[tid] = 0.0f;
  if (tid < BSEG) cnt[tid] = 0;
}

// Convert the five 256x256 fp32 weight matrices to transposed f16 (Wt[n][k]).
__global__ void k_prep(const float* __restrict__ bw1, const float* __restrict__ bw2,
                       const float* __restrict__ tw1, const float* __restrict__ tw2,
                       const float* __restrict__ ow0, _Float16* __restrict__ wt) {
  int tid = blockIdx.x * 256 + threadIdx.x;
  if (tid >= 5 * 65536) return;
  int mat = tid >> 16, idx = tid & 65535;
  int n = idx >> 8, k = idx & 255;
  const float* src = (mat == 0) ? tw1 : (mat == 1) ? tw2
                     : (mat == 2) ? bw1 : (mat == 3) ? bw2 : ow0;
  wt[mat * 65536 + n * HID + k] = (_Float16)src[k * HID + n];
}

__global__ __launch_bounds__(256) void k_count(const int* __restrict__ seg,
                                               int* __restrict__ cnt) {
  __shared__ int h[BSEG];
  const int t = threadIdx.x;
  if (t < BSEG) h[t] = 0;
  __syncthreads();
  const int base = blockIdx.x * 1024;
  for (int i = t; i < 1024; i += 256) atomicAdd(&h[seg[base + i]], 1);
  __syncthreads();
  if (t < BSEG) atomicAdd(&cnt[t], h[t]);
}

__global__ void k_segmean(const float* __restrict__ segsum,
                          const int* __restrict__ cnt,
                          float* __restrict__ segmean) {
  int tid = blockIdx.x * 256 + threadIdx.x;
  if (tid < BSEG * HID)
    segmean[tid] = segsum[tid] / fmaxf((float)cnt[tid >> 8], 1.0f);
}

// ---------- trunk: c = MLP3(nodes); segsum += c (run-length-compressed atomics) ----------
__global__ __launch_bounds__(256) void k_trunk(
    const float* __restrict__ nodes, const int* __restrict__ cseg,
    const float* __restrict__ w0, const float* __restrict__ b0,
    const _Float16* __restrict__ w1t, const float* __restrict__ b1,
    const _Float16* __restrict__ w2t, const float* __restrict__ b2,
    float* __restrict__ segsum) {
  __shared__ __align__(16) _Float16 buf[MT * LDA];
  __shared__ float xs[MT * 3];
  const int t = threadIdx.x;
  const int row0 = blockIdx.x * MT;
  if (t < MT * 3) xs[t] = nodes[row0 * 3 + t];
  __syncthreads();
  { // layer0: 3 -> 256, VALU (K too small for WMMA)
    const int n = t;
    const float wa = w0[n], wb = w0[HID + n], wc = w0[2 * HID + n];
    const float bb = b0[n];
#pragma unroll 4
    for (int r = 0; r < MT; ++r) {
      float h = fmaf(xs[r * 3 + 2], wc, fmaf(xs[r * 3 + 1], wb, fmaf(xs[r * 3 + 0], wa, bb)));
      buf[r * LDA + n] = (_Float16)fmaxf(h, 0.0f);
    }
  }
  __syncthreads();
  const int lane = t & 31, wv = t >> 5;
  const int mtile = wv >> 1, nbase = (wv & 1) * 128;
  v8f accs[8];
  // layer1 (ReLU) — compute-all, barrier, write-back into same LDS buffer
  layer_accs(buf, w1t, mtile, nbase, lane, accs);
  add_bias(accs, b1, nbase, lane);
  __syncthreads();
  store_accs(buf, accs, mtile, nbase, lane, true);
  __syncthreads();
  // layer2 (no ReLU) + segment-sum atomics (seg ids are sorted -> compress runs)
  layer_accs(buf, w2t, mtile, nbase, lane, accs);
  add_bias(accs, b2, nbase, lane);
  {
    const int rb = row0 + mtile * 16 + ((lane & 16) ? 8 : 0);
    const int n0 = nbase + (lane & 15);
    int s[8];
#pragma unroll
    for (int r = 0; r < 8; ++r) s[r] = cseg[rb + r];
#pragma unroll
    for (int nt = 0; nt < 8; ++nt) {
      const int n = n0 + nt * 16;
      float pend = accs[nt][0];
#pragma unroll
      for (int r = 1; r < 8; ++r) {
        if (s[r] != s[r - 1]) { atomicAdd(&segsum[s[r - 1] * HID + n], pend); pend = 0.0f; }
        pend += accs[nt][r];
      }
      atomicAdd(&segsum[s[7] * HID + n], pend);
    }
  }
}

// ---------- branch: f = MLP3(known); feat = f * mean[seg]; head 256->256(ReLU)->3 ----------
__global__ __launch_bounds__(256) void k_branch(
    const float* __restrict__ known, const int* __restrict__ kseg,
    const float* __restrict__ w0, const float* __restrict__ b0,
    const _Float16* __restrict__ w1t, const float* __restrict__ b1,
    const _Float16* __restrict__ w2t, const float* __restrict__ b2,
    const _Float16* __restrict__ ow0t, const float* __restrict__ ob0,
    const float* __restrict__ ow1, const float* __restrict__ ob1,
    const float* __restrict__ segmean, float* __restrict__ out) {
  __shared__ __align__(16) _Float16 buf[MT * LDA];
  __shared__ float xs[MT * 3];
  const int t = threadIdx.x;
  const int row0 = blockIdx.x * MT;
  if (t < MT * 3) xs[t] = known[row0 * 3 + t];
  __syncthreads();
  { // layer0
    const int n = t;
    const float wa = w0[n], wb = w0[HID + n], wc = w0[2 * HID + n];
    const float bb = b0[n];
#pragma unroll 4
    for (int r = 0; r < MT; ++r) {
      float h = fmaf(xs[r * 3 + 2], wc, fmaf(xs[r * 3 + 1], wb, fmaf(xs[r * 3 + 0], wa, bb)));
      buf[r * LDA + n] = (_Float16)fmaxf(h, 0.0f);
    }
  }
  __syncthreads();
  const int lane = t & 31, wv = t >> 5;
  const int mtile = wv >> 1, nbase = (wv & 1) * 128;
  v8f accs[8];
  // layer1 (ReLU)
  layer_accs(buf, w1t, mtile, nbase, lane, accs);
  add_bias(accs, b1, nbase, lane);
  __syncthreads();
  store_accs(buf, accs, mtile, nbase, lane, true);
  __syncthreads();
  // layer2 (no ReLU) then fused feat = f * segmean[seg]
  layer_accs(buf, w2t, mtile, nbase, lane, accs);
  add_bias(accs, b2, nbase, lane);
  {
    const int rb = row0 + mtile * 16 + ((lane & 16) ? 8 : 0);
    const int n0 = nbase + (lane & 15);
    int s[8];
#pragma unroll
    for (int r = 0; r < 8; ++r) s[r] = kseg[rb + r];
#pragma unroll
    for (int nt = 0; nt < 8; ++nt) {
      const int n = n0 + nt * 16;
#pragma unroll
      for (int r = 0; r < 8; ++r) accs[nt][r] *= segmean[s[r] * HID + n];
    }
  }
  __syncthreads();
  store_accs(buf, accs, mtile, nbase, lane, false);
  __syncthreads();
  // head layer (ow0, ReLU)
  layer_accs(buf, ow0t, mtile, nbase, lane, accs);
  add_bias(accs, ob0, nbase, lane);
  __syncthreads();
  store_accs(buf, accs, mtile, nbase, lane, true);
  __syncthreads();
  // final 256 -> 3 projection (tiny N: VALU dot products)
  if (t < MT * 3) {
    const int row = t / 3, col = t % 3;
    float s = ob1[col];
#pragma unroll 8
    for (int k = 0; k < HID; ++k)
      s += (float)buf[row * LDA + k] * ow1[k * 3 + col];
    out[row0 * 3 + t] = s;
  }
}

// ---------- host ----------
extern "C" void kernel_launch(void* const* d_in, const int* in_sizes, int n_in,
                              void* d_out, int out_size, void* d_ws, size_t ws_size,
                              hipStream_t stream) {
  const float* known = (const float*)d_in[0];
  const float* nodes = (const float*)d_in[1];
  const int*   kseg  = (const int*)d_in[2];
  const int*   cseg  = (const int*)d_in[3];
  const float* bw0 = (const float*)d_in[4];
  const float* bb0 = (const float*)d_in[5];
  const float* bw1 = (const float*)d_in[6];
  const float* bb1 = (const float*)d_in[7];
  const float* bw2 = (const float*)d_in[8];
  const float* bb2 = (const float*)d_in[9];
  const float* tw0 = (const float*)d_in[10];
  const float* tb0 = (const float*)d_in[11];
  const float* tw1 = (const float*)d_in[12];
  const float* tb1 = (const float*)d_in[13];
  const float* tw2 = (const float*)d_in[14];
  const float* tb2 = (const float*)d_in[15];
  const float* ow0 = (const float*)d_in[16];
  const float* ob0 = (const float*)d_in[17];
  const float* ow1 = (const float*)d_in[18];
  const float* ob1 = (const float*)d_in[19];
  float* out = (float*)d_out;

  // workspace layout: 5 x f16ᵀ 256x256 weights, segsum f32, segmean f32, cnt i32
  _Float16* tw1t = (_Float16*)d_ws;
  _Float16* tw2t = tw1t + 65536;
  _Float16* bw1t = tw2t + 65536;
  _Float16* bw2t = bw1t + 65536;
  _Float16* ow0t = bw2t + 65536;
  float* segsum  = (float*)(ow0t + 65536);
  float* segmean = segsum + BSEG * HID;
  int*   cnt     = (int*)(segmean + BSEG * HID);

  k_zero<<<64, 256, 0, stream>>>(segsum, cnt);
  k_prep<<<(5 * 65536) / 256, 256, 0, stream>>>(bw1, bw2, tw1, tw2, ow0, tw1t);
  k_count<<<NC_ / 1024, 256, 0, stream>>>(cseg, cnt);
  k_trunk<<<NC_ / MT, 256, 0, stream>>>(nodes, cseg, tw0, tb0, tw1t, tb1, tw2t, tb2, segsum);
  k_segmean<<<64, 256, 0, stream>>>(segsum, cnt, segmean);
  k_branch<<<NK_ / MT, 256, 0, stream>>>(known, kseg, bw0, bb0, bw1t, bb1, bw2t, bb2,
                                         ow0t, ob0, ow1, ob1, segmean, out);
}
